// FCNN_7095285973380
// MI455X (gfx1250) — matrix operand order repaired and checked by reading, not compiled, not run on hardware
//
#include <hip/hip_runtime.h>

typedef __attribute__((ext_vector_type(16))) _Float16 v16h;
typedef __attribute__((ext_vector_type(8)))  _Float16 v8h;
typedef __attribute__((ext_vector_type(8)))  float    v8f;
typedef __attribute__((ext_vector_type(2)))  float    v2f;
typedef __attribute__((ext_vector_type(8)))  unsigned int v8u;

#define B_    16
#define CIN   4
#define H_    128
#define W_    128
#define KIN   9
#define HID   50
#define COUT  2
#define FOUT  18     // COUT*3*3
#define FIN   324    // CIN*KIN*KIN
#define HO    120    // H - KIN + 1
#define OH    122    // output H/W

// padded LDS row strides (odd multiple of 4 dwords -> conflict-free b128 row access)
#define W1S   328    // W1^T row stride (valid k' 0..319)
#define NPS   72     // W2^T / W3^T / H row stride (valid k 0..63)
#define RRS   40     // bufR row stride (valid j 0..31)

// k-permutation for layer 1 (A and B both use it, so the GEMM result is unchanged):
//   dj<8          : k' = di*32 + dj*4 + c     (9 dense K-steps, from bufA[di][x][c])
//   dj==8, di<8   : k' = 288 + di*4 + c       (1 dense K-step, from bufR[m][di*4+c])
//   dj==8, di==8  : f32 WMMA K=4 tail over c  (exact, from bufT[c][x])
#define OFF_A 0      // bufA: 9*544   = 4896 h
#define OFF_R 4896   // bufR: 128*40  = 5120 h
#define OFF_T 10016  // bufT: 4*136   =  544 h
#define UNI   10560  // union size (H needs 128*72 = 9216 h)

__global__ void zero_out_k(float* out, int n) {
    int i = blockIdx.x * blockDim.x + threadIdx.x;
    if (i < n) out[i] = 0.0f;
}

__global__ void div_out_k(float* out, int n) {
    int i = blockIdx.x * blockDim.x + threadIdx.x;
    if (i >= n) return;
    int w = i % OH;
    int h = (i / OH) % OH;
    float ch = (h < 2) ? (float)(h + 1) : ((h > 119) ? (float)(122 - h) : 3.0f);
    float cw = (w < 2) ? (float)(w + 1) : ((w > 119) ? (float)(122 - w) : 3.0f);
    out[i] = out[i] / (ch * cw);
}

// A-fragment from a channel-interleaved panel: each of the 8 VGPRs is one aligned
// ds_load_b32 at (base + vbase + constant-offset).
__device__ __forceinline__ v16h load_a32(const _Float16* base, int vbase) {
    constexpr int aoff[8] = {0, 2, 4, 6, 16, 18, 20, 22};
    v8u au;
    #pragma unroll
    for (int v = 0; v < 8; ++v)
        au[v] = *reinterpret_cast<const unsigned int*>(base + vbase + aoff[v]);
    return __builtin_bit_cast(v16h, au);
}

// B-fragment: two 16-B-aligned v8h loads at p and p+16 (table k order)
__device__ __forceinline__ v16h load_b16(const _Float16* p) {
    v8h lo = *reinterpret_cast<const v8h*>(p);
    v8h hi = *reinterpret_cast<const v8h*>(p + 16);
    v16h r;
    #pragma unroll
    for (int e = 0; e < 8; ++e) { r[e] = lo[e]; r[e + 8] = hi[e]; }
    return r;
}

__global__ __launch_bounds__(256) void fcnn_wmma_k(
    const float* __restrict__ x,
    const float* __restrict__ W1, const float* __restrict__ b1,
    const float* __restrict__ W2, const float* __restrict__ b2,
    const float* __restrict__ W3, const float* __restrict__ b3,
    float* __restrict__ out)
{
    __shared__ __align__(32) _Float16 s_w1t[HID * W1S];   // permuted W1^T  32800 B
    __shared__ __align__(32) float    s_w1tl[CIN * HID];  // W1 rows (c,8,8) 800 B
    __shared__ __align__(32) _Float16 s_w2t[HID * NPS];   // W2^T            7200 B
    __shared__ __align__(32) _Float16 s_w3t[FOUT * NPS];  // W3^T            2592 B
    __shared__ float s_b1[64], s_b2[64], s_b3[32];
    // union: layer-1 staging panels (bufA/bufR/bufT), then H [128][NPS] f16
    __shared__ __align__(32) _Float16 s_u[UNI];           // 21120 B

    const int tid = threadIdx.x;
    const int wg  = blockIdx.x;
    const int bb  = wg / HO;     // batch
    const int y   = wg % HO;     // patch row

    // ---- stage W1 (f32 -> f16, k-permuted & transposed) ----
    for (int i = tid; i < FIN * HID; i += 256) {          // i = korig*50 + n
        int korig = i / HID, n = i - korig * HID;
        int c = korig / 81, rem = korig - c * 81;
        int di = rem / 9,  dj = rem - di * 9;
        if (dj < 8)
            s_w1t[n * W1S + di * 32 + dj * 4 + c] = (_Float16)W1[i];
        else if (di < 8)
            s_w1t[n * W1S + 288 + di * 4 + c]     = (_Float16)W1[i];
        else
            s_w1tl[c * HID + n] = W1[i];                  // exact f32 tail
    }
    for (int i = tid; i < HID * NPS; i += 256) {          // W2^T, zero-pad k>=50
        int n = i / NPS, k = i - n * NPS;
        s_w2t[i] = (k < HID) ? (_Float16)W2[k * HID + n] : (_Float16)0.0f;
    }
    for (int i = tid; i < FOUT * NPS; i += 256) {         // W3^T, zero-pad k>=50
        int n = i / NPS, k = i - n * NPS;
        s_w3t[i] = (k < HID) ? (_Float16)W3[k * FOUT + n] : (_Float16)0.0f;
    }
    if (tid < 64) {
        s_b1[tid] = (tid < HID) ? b1[tid] : 0.0f;
        s_b2[tid] = (tid < HID) ? b2[tid] : 0.0f;
    }
    if (tid < 32) s_b3[tid] = (tid < FOUT) ? b3[tid] : 0.0f;
    // ---- stage x panels ----
    for (int i = tid; i < 9 * 544; i += 256) {            // bufA[di][xx][c]
        int di = i / 544, r = i - di * 544, xx = r >> 2, c = r & 3;
        float v = (xx < W_) ? x[((bb * CIN + c) * H_ + (y + di)) * W_ + xx] : 0.0f;
        s_u[OFF_A + i] = (_Float16)v;
    }
    for (int i = tid; i < 128 * 32; i += 256) {           // bufR[m][di*4+c] = x(c, y+di, m+8)
        int m = i >> 5, j = i & 31, di = j >> 2, c = j & 3, xx = m + 8;
        float v = (xx < W_) ? x[((bb * CIN + c) * H_ + (y + di)) * W_ + xx] : 0.0f;
        s_u[OFF_R + m * RRS + j] = (_Float16)v;
    }
    for (int i = tid; i < 4 * 136; i += 256) {            // bufT[c][xx] = x(c, y+8, xx)
        int c = i / 136, xx = i - c * 136;
        float v = (xx < W_) ? x[((bb * CIN + c) * H_ + (y + 8)) * W_ + xx] : 0.0f;
        s_u[OFF_T + i] = (_Float16)v;
    }
    __syncthreads();

    const int wv   = tid >> 5;       // wave id = M-tile (0..7)
    const int lane = tid & 31;
    const int lLo  = lane & 15;
    const int lHi  = lane >> 4;
    const int mrow = wv * 16 + lLo;  // A-fragment row (= x position for layer 1)
    const int vbaseA = 4 * mrow + 8 * lHi;      // bufA: stride 4 h per x
    const int vbaseR = RRS * mrow + 8 * lHi;    // bufR
    // per-lane clamped B row indices (garbage columns are OK, just stay in-bounds)
    int nrow1[4], nrow3[2];
    #pragma unroll
    for (int nt = 0; nt < 4; ++nt) {
        int n = nt * 16 + lLo;
        nrow1[nt] = (n < HID) ? n : (HID - 1);
    }
    #pragma unroll
    for (int nt = 0; nt < 2; ++nt) {
        int n = nt * 16 + lLo;
        nrow3[nt] = (n < FOUT) ? n : (FOUT - 1);
    }

    v8f vzero;
    #pragma unroll
    for (int i = 0; i < 8; ++i) vzero[i] = 0.0f;

    // ================= layer 1: [128 x 324] @ [324 x 50] =================
    v8f acc[4];
    #pragma unroll
    for (int i = 0; i < 4; ++i) acc[i] = vzero;

    // software-pipelined: prefetch step st+1 while WMMAs of step st execute
    v16h aC = load_a32(s_u + OFF_A, vbaseA);
    v16h bC[4];
    #pragma unroll
    for (int nt = 0; nt < 4; ++nt)
        bC[nt] = load_b16(&s_w1t[nrow1[nt] * W1S + (lHi << 3)]);

    #pragma unroll
    for (int st = 0; st < 10; ++st) {        // steps 0..8: bufA; step 9: bufR
        v16h aN, bN[4];
        if (st < 9) {
            aN = (st == 8) ? load_a32(s_u + OFF_R, vbaseR)
                           : load_a32(s_u + OFF_A + 544 * (st + 1), vbaseA);
            #pragma unroll
            for (int nt = 0; nt < 4; ++nt)
                bN[nt] = load_b16(&s_w1t[nrow1[nt] * W1S + (st + 1) * 32 + (lHi << 3)]);
        }
        #pragma unroll
        for (int nt = 0; nt < 4; ++nt)
            acc[nt] = __builtin_amdgcn_wmma_f32_16x16x32_f16(
                false, aC, false, bC[nt], (short)0, acc[nt], false, false);
        if (st < 9) {
            aC = aN;
            #pragma unroll
            for (int nt = 0; nt < 4; ++nt) bC[nt] = bN[nt];
        }
    }
    {   // exact f32 tail: (c, di=8, dj=8), K = 4 over c
        v2f a2;
        #pragma unroll
        for (int e = 0; e < 2; ++e)          // k_local = c = e + 2*lHi
            a2[e] = (float)s_u[OFF_T + (e + 2 * lHi) * 136 + mrow + 8];
        #pragma unroll
        for (int nt = 0; nt < 4; ++nt) {
            v2f b2v;
            #pragma unroll
            for (int e = 0; e < 2; ++e)
                b2v[e] = s_w1tl[(e + 2 * lHi) * HID + nrow1[nt]];
            acc[nt] = __builtin_amdgcn_wmma_f32_16x16x4_f32(
                false, a2, false, b2v, (short)0, acc[nt], false, false);
        }
    }
    __syncthreads();  // all panel reads done; s_u becomes hidden buffer
    #pragma unroll
    for (int nt = 0; nt < 4; ++nt) {
        const int n = nt * 16 + lLo;
        #pragma unroll
        for (int r = 0; r < 8; ++r) {
            const int M = wv * 16 + r + 8 * lHi;
            float v = acc[nt][r] + s_b1[n];
            s_u[M * NPS + n] = (_Float16)fmaxf(v, 0.0f);  // cols >=50: garbage, OK
        }
    }
    __syncthreads();

    // ================= layer 2: [128 x 50] @ [50 x 50] =================
    {
        v16h a2c[2], b2f[2][4];
        #pragma unroll
        for (int ks = 0; ks < 2; ++ks) {
            const int kb = ks * 32;
            const v8h alo = *reinterpret_cast<const v8h*>(&s_u[mrow * NPS + kb + (lHi << 3)]);
            const v8h ahi = *reinterpret_cast<const v8h*>(&s_u[mrow * NPS + kb + 16 + (lHi << 3)]);
            #pragma unroll
            for (int e = 0; e < 8; ++e) { a2c[ks][e] = alo[e]; a2c[ks][e + 8] = ahi[e]; }
            #pragma unroll
            for (int nt = 0; nt < 4; ++nt)
                b2f[ks][nt] = load_b16(&s_w2t[nrow1[nt] * NPS + kb + (lHi << 3)]);
        }
        #pragma unroll
        for (int i = 0; i < 4; ++i) acc[i] = vzero;
        #pragma unroll
        for (int ks = 0; ks < 2; ++ks)
            #pragma unroll
            for (int nt = 0; nt < 4; ++nt)
                acc[nt] = __builtin_amdgcn_wmma_f32_16x16x32_f16(
                    false, a2c[ks], false, b2f[ks][nt], (short)0, acc[nt], false, false);
    }
    __syncthreads();  // all H1 reads done
    #pragma unroll
    for (int nt = 0; nt < 4; ++nt) {
        const int n = nt * 16 + lLo;
        #pragma unroll
        for (int r = 0; r < 8; ++r) {
            const int M = wv * 16 + r + 8 * lHi;
            float v = acc[nt][r] + s_b2[n];
            s_u[M * NPS + n] = (_Float16)fmaxf(v, 0.0f);
        }
    }
    __syncthreads();

    // ================= layer 3: [128 x 50] @ [50 x 18] =================
    v8f acc3[2];
    acc3[0] = vzero; acc3[1] = vzero;
    {
        v16h a3c[2], b3f[2][2];
        #pragma unroll
        for (int ks = 0; ks < 2; ++ks) {
            const int kb = ks * 32;
            const v8h alo = *reinterpret_cast<const v8h*>(&s_u[mrow * NPS + kb + (lHi << 3)]);
            const v8h ahi = *reinterpret_cast<const v8h*>(&s_u[mrow * NPS + kb + 16 + (lHi << 3)]);
            #pragma unroll
            for (int e = 0; e < 8; ++e) { a3c[ks][e] = alo[e]; a3c[ks][e + 8] = ahi[e]; }
            #pragma unroll
            for (int nt = 0; nt < 2; ++nt)
                b3f[ks][nt] = load_b16(&s_w3t[nrow3[nt] * NPS + kb + (lHi << 3)]);
        }
        #pragma unroll
        for (int ks = 0; ks < 2; ++ks)
            #pragma unroll
            for (int nt = 0; nt < 2; ++nt)
                acc3[nt] = __builtin_amdgcn_wmma_f32_16x16x32_f16(
                    false, a3c[ks], false, b3f[ks][nt], (short)0, acc3[nt], false, false);
    }

    // ================= fold: overlap-add 3x3 patches via global atomics =================
    #pragma unroll
    for (int nt = 0; nt < 2; ++nt) {
        const int col = nt * 16 + lLo;           // c*9 + ki*3 + kj
        if (col < FOUT) {
            const int c   = col / 9;
            const int rem = col - 9 * c;
            const int pi  = rem / 3;
            const int pj  = rem - 3 * pi;
            const float bias = s_b3[col];
            #pragma unroll
            for (int r = 0; r < 8; ++r) {
                const int M = wv * 16 + r + 8 * lHi;   // = x position
                if (M < HO) {
                    float v = acc3[nt][r] + bias;
                    atomicAdd(&out[((bb * COUT + c) * OH + (y + pi)) * OH + (M + pj)], v);
                }
            }
        }
    }
}

extern "C" void kernel_launch(void* const* d_in, const int* in_sizes, int n_in,
                              void* d_out, int out_size, void* d_ws, size_t ws_size,
                              hipStream_t stream) {
    (void)in_sizes; (void)n_in; (void)d_ws; (void)ws_size;
    const float* x  = (const float*)d_in[0];
    const float* W1 = (const float*)d_in[1];
    const float* b1 = (const float*)d_in[2];
    const float* W2 = (const float*)d_in[3];
    const float* b2 = (const float*)d_in[4];
    const float* W3 = (const float*)d_in[5];
    const float* b3 = (const float*)d_in[6];
    float* out = (float*)d_out;

    zero_out_k<<<(out_size + 255) / 256, 256, 0, stream>>>(out, out_size);
    fcnn_wmma_k<<<B_ * HO, 256, 0, stream>>>(x, W1, b1, W2, b2, W3, b3, out);
    div_out_k<<<(out_size + 255) / 256, 256, 0, stream>>>(out, out_size);
}
